// SparseAE_CNN_Mamba_30253749633168
// MI455X (gfx1250) — matrix operand order repaired and checked by reading, not compiled, not run on hardware
//
#include <hip/hip_runtime.h>
#include <hip/hip_bf16.h>

// ---------------- problem constants ----------------
#define BB    128
#define LL    512
#define VV    30522
#define EE    256
#define LATD  128
#define CCH   128
#define NCLS  4
#define DIN   256
#define DST   64
#define DCONV 4
#define DTR   8
#define PO    64              // pooled sequence length (L / 8)
#define DBCW  (DTR + 2*DST)   // 136
#define SPW   0.001f

typedef __bf16 bf16;
typedef bf16  v16bf __attribute__((ext_vector_type(16)));
typedef bf16  v8bf  __attribute__((ext_vector_type(8)));
typedef float v8f   __attribute__((ext_vector_type(8)));

// ---------------- WMMA fragment helpers ----------------
// A-fragment (16x32 bf16, MxK): lane = half*16 + m; K = half*16..half*16+15 contiguous
// -> two adjacent 16B loads.
// B-fragment (32x16 bf16, KxN) from W[N,K] row-major: lane holds column n = lane&15,
// K = half*16..half*16+15 contiguous -> two adjacent 16B loads from row n of W.

__device__ __forceinline__ v16bf cat8(v8bf lo, v8bf hi) {
  return __builtin_shufflevector(lo, hi, 0,1,2,3,4,5,6,7,8,9,10,11,12,13,14,15);
}
__device__ __forceinline__ v16bf zero16() { v16bf z = {}; return z; }

__device__ __forceinline__ v16bf load_a_frag(const bf16* __restrict__ row, int k0, int half) {
  const bf16* p = row + k0 + half * 16;
  return cat8(*(const v8bf*)p, *(const v8bf*)(p + 8));
}
__device__ __forceinline__ v16bf load_b_frag(const bf16* __restrict__ row, int k0, int half) {
  const bf16* p = row + k0 + half * 16;
  return cat8(*(const v8bf*)p, *(const v8bf*)(p + 8));
}

// ---------------- generic bf16 WMMA GEMM, 16x(16*NT) tile per wave ----------------
// D[M,N] = act( A[M,K] @ W[N,K]^T + bias ). A fragment is loaded ONCE per K-step and
// reused for NT back-to-back WMMAs -> streamed activations hit HBM exactly once
// (weights stay L2-resident). rowIdx != null fuses the embedding gather.
template <int NT>
__global__ void __launch_bounds__(32)
gemm_wmma(const bf16* __restrict__ A, int lda, const int* __restrict__ rowIdx,
          const bf16* __restrict__ W, int ldw,
          const float* __restrict__ bias,
          float* __restrict__ Cf, bf16* __restrict__ Cb, int ldc,
          int M, int N, int K, int relu, float* __restrict__ spAcc)
{
  const int lane = threadIdx.x;
  const int mrow = lane & 15;
  const int half = lane >> 4;
  const int m = blockIdx.y * 16 + mrow;

  long ar = rowIdx ? (long)rowIdx[m] : (long)m;
  const bf16* Arow = A + ar * (long)lda;

  const bf16* Wrow[NT];
  bool nvalid[NT];
#pragma unroll
  for (int nt = 0; nt < NT; ++nt) {
    const int n = (blockIdx.x * NT + nt) * 16 + mrow;
    nvalid[nt] = (n < N);
    Wrow[nt] = W + (long)n * (long)ldw;
  }

  v8f acc[NT];
#pragma unroll
  for (int nt = 0; nt < NT; ++nt) { v8f z = {}; acc[nt] = z; }

  for (int k0 = 0; k0 < K; k0 += 32) {
    if (k0 + 32 < K) __builtin_prefetch(Arow + k0 + 32, 0, 3);
    const v16bf af = load_a_frag(Arow, k0, half);
#pragma unroll
    for (int nt = 0; nt < NT; ++nt) {
      v16bf bf = nvalid[nt] ? load_b_frag(Wrow[nt], k0, half) : zero16();
      acc[nt] = __builtin_amdgcn_wmma_f32_16x16x32_bf16(false, af, false, bf,
                                                        (short)0, acc[nt], false, false);
    }
  }

  // C/D layout: lane j<16 -> rows m0..m0+7, col n0+j ; lane j>=16 -> rows m0+8..m0+15.
  const int rbase = blockIdx.y * 16 + half * 8;
  float ssum = 0.f;
#pragma unroll
  for (int nt = 0; nt < NT; ++nt) {
    const int ncol = (blockIdx.x * NT + nt) * 16 + (lane & 15);
    if (ncol < N) {
      const float bv = bias ? bias[ncol] : 0.f;
#pragma unroll
      for (int r = 0; r < 8; ++r) {
        float v = acc[nt][r] + bv;
        if (relu) v = v > 0.f ? v : 0.f;
        ssum += fabsf(v);
        long idx = (long)(rbase + r) * ldc + ncol;
        if (Cf) Cf[idx] = v;
        if (Cb) Cb[idx] = (bf16)v;
      }
    }
  }
  if (spAcc) {
#pragma unroll
    for (int off = 16; off > 0; off >>= 1) ssum += __shfl_xor(ssum, off, 32);
    if (lane == 0) atomicAdd(spAcc, ssum);
  }
}

// ---------------- conv(width 5) + relu + 8:1 max-pool, WMMA ----------------
// One wave: 16 l-positions x all 128 channels (8 accumulators, 160 WMMAs).
// latent rows are loaded once; each lane's 8 accumulator lanes are one pool window.
__global__ void __launch_bounds__(32)
conv_pool_wmma(const bf16* __restrict__ latent, const bf16* __restrict__ Wt,
               const float* __restrict__ bias, bf16* __restrict__ featp)
{
  const int lane = threadIdx.x;
  const int mrow = lane & 15, half = lane >> 4;
  const int lt = blockIdx.x;          // l tile 0..31
  const int b  = blockIdx.y;
  const int l  = lt * 16 + mrow;      // sequence pos (A-fragment row)

  v8f acc[8];
#pragma unroll
  for (int nt = 0; nt < 8; ++nt) { v8f z = {}; acc[nt] = z; }

  for (int t = 0; t < 5; ++t) {
    const int ls = l + t - 2;
    const bool valid = (ls >= 0) && (ls < LL);
    const bf16* arow = latent + ((long)b * LL + ls) * LATD;
#pragma unroll
    for (int k0 = 0; k0 < LATD; k0 += 32) {
      const v16bf af = valid ? load_a_frag(arow, k0, half) : zero16();
#pragma unroll
      for (int nt = 0; nt < 8; ++nt) {
        const bf16* wrow = Wt + ((long)t * CCH + nt * 16 + mrow) * LATD;
        v16bf bf = load_b_frag(wrow, k0, half);
        acc[nt] = __builtin_amdgcn_wmma_f32_16x16x32_bf16(false, af, false, bf,
                                                          (short)0, acc[nt], false, false);
      }
    }
  }

  const int p = lt * 2 + half;        // pool window index
#pragma unroll
  for (int nt = 0; nt < 8; ++nt) {
    const float bv = bias[nt * 16 + (lane & 15)];
    float mx = 0.f;                   // relu outputs are >= 0
#pragma unroll
    for (int r = 0; r < 8; ++r) {
      float v = acc[nt][r] + bv;
      v = v > 0.f ? v : 0.f;
      mx = v > mx ? v : mx;
    }
    featp[((long)b * PO + p) * CCH + nt * 16 + (lane & 15)] = (bf16)mx;
  }
}

// ---------------- small / elementwise kernels ----------------
__global__ void cvt_f32_bf16(const float* __restrict__ s, bf16* __restrict__ d, int n) {
  int i = blockIdx.x * blockDim.x + threadIdx.x;
  if (i < n) d[i] = (bf16)s[i];
}

__global__ void conv_repack(const float* __restrict__ W, bf16* __restrict__ Wt) {
  int i = blockIdx.x * blockDim.x + threadIdx.x;      // (t, c, i) with i fastest
  if (i >= 5 * CCH * LATD) return;
  int idx = i;
  int ic = idx % LATD; idx /= LATD;
  int c  = idx % CCH;
  int t  = idx / CCH;
  Wt[i] = (bf16)W[(long)c * LATD * 5 + (long)ic * 5 + t];
}

// depthwise causal conv (k=4) + SiLU on xm = xz[:, :256]
__global__ void dwconv_silu(const float* __restrict__ xz, const float* __restrict__ cw,
                            const float* __restrict__ cb,
                            float* __restrict__ u, bf16* __restrict__ ub) {
  int i = blockIdx.x * blockDim.x + threadIdx.x;
  if (i >= BB * PO * DIN) return;
  int d = i % DIN;
  int row = i / DIN;
  int p = row % PO, b = row / PO;
  float acc = cb[d];
#pragma unroll
  for (int t = 0; t < DCONV; ++t) {
    int ps = p + t - (DCONV - 1);
    if (ps >= 0) acc += xz[((long)b * PO + ps) * (2 * DIN) + d] * cw[d * DCONV + t];
  }
  float s = acc / (1.f + __expf(-acc));
  u[i] = s;
  ub[i] = (bf16)s;
}

// delta = softplus(dt @ dt_proj_W^T + b)  (K = 8, pure VALU)
__global__ void dt_delta(const float* __restrict__ dbc, const float* __restrict__ dtW,
                         const float* __restrict__ dtb, float* __restrict__ delta) {
  int i = blockIdx.x * blockDim.x + threadIdx.x;
  if (i >= BB * PO * DIN) return;
  int d = i % DIN;
  long row = i / DIN;
  float a = dtb[d];
#pragma unroll
  for (int r = 0; r < DTR; ++r) a += dbc[row * DBCW + r] * dtW[d * DTR + r];
  delta[i] = (a > 20.f) ? a : log1pf(__expf(a));
}

// selective scan: thread = (b, d), h[64] + A[64] register resident, B/C via LDS
__global__ void __launch_bounds__(256)
mamba_scan(const float* __restrict__ dbc, const float* __restrict__ delta,
           const float* __restrict__ u, const float* __restrict__ Alog,
           float* __restrict__ y)
{
  __shared__ float sB[DST];
  __shared__ float sC[DST];
  const int b = blockIdx.x;
  const int d = threadIdx.x;
  float An[DST], h[DST];
#pragma unroll
  for (int n = 0; n < DST; ++n) { An[n] = -__expf(Alog[d * DST + n]); h[n] = 0.f; }
  for (int p = 0; p < PO; ++p) {
    const long row = (long)b * PO + p;
    if (d < DST) {
      sB[d] = dbc[row * DBCW + DTR + d];
      sC[d] = dbc[row * DBCW + DTR + DST + d];
    }
    __syncthreads();
    const float dl = delta[row * DIN + d];
    const float du = dl * u[row * DIN + d];
    float yy = 0.f;
#pragma unroll
    for (int n = 0; n < DST; ++n) {
      float dA = __expf(dl * An[n]);
      h[n] = dA * h[n] + du * sB[n];
      yy += h[n] * sC[n];
    }
    y[row * DIN + d] = yy;
    __syncthreads();
  }
}

// ybar = (y + u*D) * silu(z), emitted bf16 for the out_proj WMMA GEMM
__global__ void mamba_gate(const float* __restrict__ y, const float* __restrict__ u,
                           const float* __restrict__ Dp, const float* __restrict__ xz,
                           bf16* __restrict__ yb) {
  int i = blockIdx.x * blockDim.x + threadIdx.x;
  if (i >= BB * PO * DIN) return;
  int d = i % DIN;
  long row = i / DIN;
  float z = xz[row * (2 * DIN) + DIN + d];
  float g = z / (1.f + __expf(-z));
  yb[i] = (bf16)((y[i] + u[i] * Dp[d]) * g);
}

__global__ void mean_pool(const float* __restrict__ mo, float* __restrict__ pooled) {
  int i = blockIdx.x * blockDim.x + threadIdx.x;
  if (i >= BB * CCH) return;
  int c = i % CCH, b = i / CCH;
  float s = 0.f;
  for (int p = 0; p < PO; ++p) s += mo[((long)b * PO + p) * CCH + c];
  pooled[i] = s * (1.f / PO);
}

__global__ void fc_out(const float* __restrict__ pooled, const float* __restrict__ fcW,
                       const float* __restrict__ fcb, float* __restrict__ out) {
  int i = blockIdx.x * blockDim.x + threadIdx.x;
  if (i >= BB * NCLS) return;
  int k = i % NCLS, b = i / NCLS;
  float s = fcb[k];
  for (int c = 0; c < CCH; ++c) s += pooled[b * CCH + c] * fcW[k * CCH + c];
  out[i] = s;
}

__global__ void zero1(float* p) { if (threadIdx.x == 0 && blockIdx.x == 0) *p = 0.f; }

__global__ void fin_sparsity(const float* __restrict__ acc, float* __restrict__ dst) {
  *dst = SPW * (*acc) / (float)((long)BB * LL * LATD);
}

// ---------------- host orchestration ----------------
extern "C" void kernel_launch(void* const* d_in, const int* in_sizes, int n_in,
                              void* d_out, int out_size, void* d_ws, size_t ws_size,
                              hipStream_t stream)
{
  (void)in_sizes; (void)n_in; (void)out_size; (void)ws_size;
  const int*   tokens   = (const int*)  d_in[0];
  const float* embedW   = (const float*)d_in[1];
  const float* encW     = (const float*)d_in[2];
  const float* encb     = (const float*)d_in[3];
  const float* decW     = (const float*)d_in[4];
  const float* decb     = (const float*)d_in[5];
  const float* convW    = (const float*)d_in[6];
  const float* convb    = (const float*)d_in[7];
  const float* inprojW  = (const float*)d_in[8];
  const float* c1W      = (const float*)d_in[9];
  const float* c1b      = (const float*)d_in[10];
  const float* xprojW   = (const float*)d_in[11];
  const float* dtW      = (const float*)d_in[12];
  const float* dtb      = (const float*)d_in[13];
  const float* Alog     = (const float*)d_in[14];
  const float* Dp       = (const float*)d_in[15];
  const float* outprojW = (const float*)d_in[16];
  const float* fcW      = (const float*)d_in[17];
  const float* fcb      = (const float*)d_in[18];

  char* ws = (char*)d_ws;
  size_t off = 0;
  auto carve = [&](size_t bytes) -> char* {
    char* p = ws + off;
    off += bytes;
    off = (off + 255) & ~(size_t)255;
    return p;
  };

  const long MT = (long)BB * LL;   // 65536 token rows
  const long MP = (long)BB * PO;   // 8192 pooled rows

  bf16*  embedB   = (bf16*) carve((size_t)VV * EE * 2);
  bf16*  encWb    = (bf16*) carve((size_t)LATD * EE * 2);
  bf16*  decWb    = (bf16*) carve((size_t)EE * LATD * 2);
  bf16*  convWtb  = (bf16*) carve((size_t)5 * CCH * LATD * 2);
  bf16*  inprojWb = (bf16*) carve((size_t)2 * DIN * CCH * 2);
  bf16*  xprojWb  = (bf16*) carve((size_t)DBCW * DIN * 2);
  bf16*  outprojWb= (bf16*) carve((size_t)CCH * DIN * 2);
  bf16*  latentB  = (bf16*) carve((size_t)MT * LATD * 2);
  bf16*  featpB   = (bf16*) carve((size_t)MP * CCH * 2);
  float* xz       = (float*)carve((size_t)MP * 2 * DIN * 4);
  float* uF       = (float*)carve((size_t)MP * DIN * 4);
  bf16*  uB       = (bf16*) carve((size_t)MP * DIN * 2);
  float* dbc      = (float*)carve((size_t)MP * DBCW * 4);
  float* deltaF   = (float*)carve((size_t)MP * DIN * 4);
  float* yF       = (float*)carve((size_t)MP * DIN * 4);
  bf16*  ybarB    = (bf16*) carve((size_t)MP * DIN * 2);
  float* moF      = (float*)carve((size_t)MP * CCH * 4);
  float* pooled   = (float*)carve((size_t)BB * CCH * 4);
  float* spAcc    = (float*)carve(256);

  float* outF   = (float*)d_out;
  float* recon  = outF + BB * NCLS;                       // (B*L, E) f32
  float* sparsO = outF + BB * NCLS + MT * EE;             // scalar

  // weight conversions (deterministic, every launch)
  {
    int n;
    n = VV * EE;        cvt_f32_bf16<<<(n + 255) / 256, 256, 0, stream>>>(embedW,   embedB,    n);
    n = LATD * EE;      cvt_f32_bf16<<<(n + 255) / 256, 256, 0, stream>>>(encW,     encWb,     n);
    n = EE * LATD;      cvt_f32_bf16<<<(n + 255) / 256, 256, 0, stream>>>(decW,     decWb,     n);
    n = 2 * DIN * CCH;  cvt_f32_bf16<<<(n + 255) / 256, 256, 0, stream>>>(inprojW,  inprojWb,  n);
    n = DBCW * DIN;     cvt_f32_bf16<<<(n + 255) / 256, 256, 0, stream>>>(xprojW,   xprojWb,   n);
    n = CCH * DIN;      cvt_f32_bf16<<<(n + 255) / 256, 256, 0, stream>>>(outprojW, outprojWb, n);
    n = 5 * CCH * LATD; conv_repack <<<(n + 255) / 256, 256, 0, stream>>>(convW, convWtb);
  }
  zero1<<<1, 1, 0, stream>>>(spAcc);

  // 1) encoder: latent = relu(embed[tokens] @ encW^T + b); fused gather + sparsity sum
  //    N = 128 -> one 16x128 supertile per wave: each embed row read exactly once.
  gemm_wmma<8><<<dim3(1, MT / 16), 32, 0, stream>>>(
      embedB, EE, tokens, encWb, EE, encb,
      nullptr, latentB, LATD, (int)MT, LATD, EE, 1, spAcc);

  // 2) decoder: recon = latent @ decW^T + b  (f32 straight into d_out)
  gemm_wmma<8><<<dim3(2, MT / 16), 32, 0, stream>>>(
      latentB, LATD, nullptr, decWb, LATD, decb,
      recon, nullptr, EE, (int)MT, EE, LATD, 0, nullptr);

  // 3) conv(5) + relu + 8:1 max-pool (one wave = 16 l x 128 c, latent read once)
  conv_pool_wmma<<<dim3(LL / 16, BB), 32, 0, stream>>>(latentB, convWtb, convb, featpB);

  // 4) mamba in_proj: xz = feat @ inprojW^T   (N = 512 -> 4 supertiles)
  gemm_wmma<8><<<dim3(4, MP / 16), 32, 0, stream>>>(
      featpB, CCH, nullptr, inprojWb, CCH, nullptr,
      xz, nullptr, 2 * DIN, (int)MP, 2 * DIN, CCH, 0, nullptr);

  // 5) depthwise conv + SiLU -> u
  int nU = (int)(MP * DIN);
  dwconv_silu<<<(nU + 255) / 256, 256, 0, stream>>>(xz, c1W, c1b, uF, uB);

  // 6) x_proj: dbc = u @ xprojW^T   (N = 136, edge-guarded)
  gemm_wmma<8><<<dim3(2, MP / 16), 32, 0, stream>>>(
      uB, DIN, nullptr, xprojWb, DIN, nullptr,
      dbc, nullptr, DBCW, (int)MP, DBCW, DIN, 0, nullptr);

  // 7) delta = softplus(dt @ dtW^T + b)   (K = 8, VALU)
  dt_delta<<<(nU + 255) / 256, 256, 0, stream>>>(dbc, dtW, dtb, deltaF);

  // 8) selective scan (sequential over 64 steps, register-resident state)
  mamba_scan<<<BB, DIN, 0, stream>>>(dbc, deltaF, uF, Alog, yF);

  // 9) gating: ybar = (y + u*D) * silu(z)
  mamba_gate<<<(nU + 255) / 256, 256, 0, stream>>>(yF, uF, Dp, xz, ybarB);

  // 10) out_proj: mo = ybar @ outprojW^T
  gemm_wmma<8><<<dim3(1, MP / 16), 32, 0, stream>>>(
      ybarB, DIN, nullptr, outprojWb, DIN, nullptr,
      moF, nullptr, CCH, (int)MP, CCH, DIN, 0, nullptr);

  // 11) heads
  mean_pool<<<(BB * CCH + 255) / 256, 256, 0, stream>>>(moF, pooled);
  fc_out<<<(BB * NCLS + 255) / 256, 256, 0, stream>>>(pooled, fcW, fcb, outF);
  fin_sparsity<<<1, 1, 0, stream>>>(spAcc, sparsO);
}
